// AttentionModule_60799557042445
// MI455X (gfx1250) — matrix operand order repaired and checked by reading, not compiled, not run on hardware
//
#include <hip/hip_runtime.h>
#include <hip/hip_bf16.h>
#include <math.h>

// ---------------------------------------------------------------------------
// Types for CDNA5 WMMA
// ---------------------------------------------------------------------------
typedef __attribute__((ext_vector_type(16))) __bf16 v16bf;
typedef __attribute__((ext_vector_type(8)))  float  v8f;

static __device__ __forceinline__ __bf16 f2bf(float f) {
    union { float f; unsigned int u; } x; x.f = f;
    unsigned int r = x.u + 0x7FFFu + ((x.u >> 16) & 1u);   // round-to-nearest-even
    unsigned short h = (unsigned short)(r >> 16);
    __bf16 out;
    __builtin_memcpy(&out, &h, sizeof(out));
    return out;
}

// ---------------------------------------------------------------------------
// Async global -> LDS 16-byte copy (CDNA5 GLOBAL_LOAD_ASYNC_TO_LDS_B128,
// tracked by ASYNCcnt).  lds_off: byte offset of destination inside LDS
// (low 32 bits of a flat pointer to a __shared__ object).
// ---------------------------------------------------------------------------
static __device__ __forceinline__ void async_copy16(unsigned lds_off, const void* g) {
    unsigned long long ga = (unsigned long long)g;
    asm volatile("global_load_async_to_lds_b128 %0, %1, off"
                 :: "v"(lds_off), "v"(ga) : "memory");
}
static __device__ __forceinline__ void wait_async0() {
    asm volatile("s_wait_asynccnt 0" ::: "memory");
}

// ---------------------------------------------------------------------------
// Fragment loaders from LDS tiles with row stride LDSW (=40 bf16 = 80B:
// 16B-aligned and l*20 mod 64 is a bank permutation -> conflict-free b128s).
//
// A 16x32 bf16 fragment (ISA 7.12.2): lane l<16 holds row M=l, K = {0..7,
// 16..23}; lane l+16 holds row M=l, K = {8..15, 24..31}.
// ---------------------------------------------------------------------------
#define LDSW 40

static __device__ __forceinline__ v16bf frag_a(const __bf16* __restrict__ base, int lane) {
    const int l  = lane & 15;
    const int hi = lane >> 4;
    const __bf16* p = base + l * LDSW + hi * 8;
    union { v16bf v; uint4 q[2]; } u;
    u.q[0] = *(const uint4*)(p);        // K offsets +0..7
    u.q[1] = *(const uint4*)(p + 16);   // K offsets +16..23
    return u.v;
}

// B 32x16 bf16 fragment: lane l<16 holds column N=l, K = {0..7, 16..23};
// lane l+16 holds column N=l, K = {8..15, 24..31}.
static __device__ __forceinline__ v16bf frag_b(const __bf16* __restrict__ base, int lane) {
    const int l  = lane & 15;
    const int hi = lane >> 4;
    const __bf16* p = base + l * LDSW + hi * 8;
    union { v16bf v; uint4 q[2]; } u;
    u.q[0] = *(const uint4*)(p);
    u.q[1] = *(const uint4*)(p + 16);
    return u.v;
}

// ---------------------------------------------------------------------------
// Universal bf16 WMMA GEMM:  C[b] = A[b] (MxK) * Bt[b]^T (KxN)  (+ epilogue)
// Block = 256 threads = 8 waves arranged 2x4; each wave owns a 64x64 patch
// (4x4 tiles of 16x16) -> workgroup tile 128 x 256, k-chunk 32.
// A (128x32) and B (256x32) slabs are staged into LDS via async DMA with
// double buffering; all 8 fragments are fetched with ds_load_b128 before the
// 16-WMMA burst so LDS latency is covered by partial dscnt waits.
// MODE 0: Cb = bf16(acc + bias[col])                       (phi)
// MODE 1: Cb = bf16(sigmoid(acc + bias[col]))              (theta)
// MODE 2: Cf = acc                                         (S scores)
// MODE 3: Cf = acc + resid                                 (weighted + x)
// ---------------------------------------------------------------------------
template <int MODE>
__global__ __launch_bounds__(256) void gemm_bf16_wmma(
    const __bf16* __restrict__ A, const __bf16* __restrict__ Bt,
    float* __restrict__ Cf, __bf16* __restrict__ Cb,
    const float* __restrict__ bias, const float* __restrict__ resid,
    int M, int Ncols, int K,
    long long strideA, long long strideB, long long strideC, long long strideR)
{
    // [2 buffers][A: rows 0..127 | B: rows 128..383][LDSW bf16 per row]
    __shared__ alignas(16) __bf16 smem[2][384 * LDSW];

    const int b    = blockIdx.z;
    const __bf16* Ab = A  + (long long)b * strideA;
    const __bf16* Bb = Bt + (long long)b * strideB;

    const int t    = threadIdx.x;
    const int lane = t & 31;
    const int wave = t >> 5;
    const int wr   = wave >> 2;          // 0..1
    const int wc   = wave & 3;           // 0..3
    const int rowWG = blockIdx.y * 128;
    const int colWG = blockIdx.x * 256;
    const int row0  = rowWG + wr * 64;

    v8f zero = {};
    v8f acc[4][4];
#pragma unroll
    for (int i = 0; i < 4; ++i)
#pragma unroll
        for (int j = 0; j < 4; ++j) acc[i][j] = zero;

    // ---- async stage of one 32-deep k-slab into smem[buf] ----
    auto stage = [&](int k, int buf) {
        __bf16* sb = &smem[buf][0];
#pragma unroll
        for (int i = 0; i < 2; ++i) {   // A tile: 128 rows x 4 chunks -> 2/thread
            const int c = t + i * 256;
            const int r = c >> 2, s = c & 3;
            const __bf16* g = Ab + (long long)(rowWG + r) * K + k + s * 8;
            unsigned lo = (unsigned)(unsigned long long)(sb + r * LDSW + s * 8);
            async_copy16(lo, g);
        }
#pragma unroll
        for (int i = 0; i < 4; ++i) {   // B tile: 256 rows x 4 chunks -> 4/thread
            const int c = t + i * 256;
            const int r = c >> 2, s = c & 3;
            const __bf16* g = Bb + (long long)(colWG + r) * K + k + s * 8;
            unsigned lo = (unsigned)(unsigned long long)(sb + (128 + r) * LDSW + s * 8);
            async_copy16(lo, g);
        }
    };

    const int nk = K >> 5;
    stage(0, 0);
    wait_async0();
    __syncthreads();

    for (int kb = 0; kb < nk; ++kb) {
        const int cur = kb & 1;
        if (kb + 1 < nk) stage((kb + 1) << 5, cur ^ 1);   // overlap DMA w/ compute

        const __bf16* sA = &smem[cur][0];
        const __bf16* sB = &smem[cur][128 * LDSW];

        // Hoist ALL fragment loads before the WMMA burst (batched ds_load_b128)
        v16bf af[4], bf[4];
#pragma unroll
        for (int i = 0; i < 4; ++i)
            af[i] = frag_a(sA + (wr * 64 + i * 16) * LDSW, lane);
#pragma unroll
        for (int j = 0; j < 4; ++j)
            bf[j] = frag_b(sB + (wc * 64 + j * 16) * LDSW, lane);

#pragma unroll
        for (int i = 0; i < 4; ++i)
#pragma unroll
            for (int j = 0; j < 4; ++j)
                acc[i][j] = __builtin_amdgcn_wmma_f32_16x16x32_bf16(
                    false, af[i], false, bf[j], (short)0, acc[i][j], false, false);

        wait_async0();     // next slab landed
        __syncthreads();   // everyone done reading cur before it is rewritten
    }

    // Epilogue.  C 16x16 f32 layout: lane l<16 -> (M = vgpr + 8*hi, N = l).
    const int l  = lane & 15;
    const int hi = lane >> 4;
#pragma unroll
    for (int i = 0; i < 4; ++i) {
#pragma unroll
        for (int j = 0; j < 4; ++j) {
            const int col = colWG + wc * 64 + j * 16 + l;
            const float bval = (MODE == 0 || MODE == 1) ? bias[col] : 0.0f;
#pragma unroll
            for (int r = 0; r < 8; ++r) {
                const int row = row0 + i * 16 + hi * 8 + r;
                float v = acc[i][j][r] + bval;
                const long long idx =
                    (long long)b * strideC + (long long)row * Ncols + col;
                if (MODE == 0) {
                    Cb[idx] = f2bf(v);
                } else if (MODE == 1) {
                    Cb[idx] = f2bf(1.0f / (1.0f + __expf(-v)));
                } else if (MODE == 2) {
                    Cf[idx] = v;
                } else {
                    const long long ridx =
                        (long long)b * strideR + (long long)row * Ncols + col;
                    Cf[idx] = v + resid[ridx];
                }
            }
        }
    }
}

// ---------------------------------------------------------------------------
// f32 -> bf16 bulk convert
// ---------------------------------------------------------------------------
__global__ void f32_to_bf16(const float* __restrict__ in, __bf16* __restrict__ out,
                            long long n) {
    long long i = (long long)blockIdx.x * blockDim.x + threadIdx.x;
    const long long stride = (long long)gridDim.x * blockDim.x;
    for (; i < n; i += stride) out[i] = f2bf(in[i]);
}

// ---------------------------------------------------------------------------
// W [rows x cols] f32  ->  WT [cols x rows] bf16  (32x32 LDS tile transpose)
// ---------------------------------------------------------------------------
__global__ __launch_bounds__(256) void transpose_conv(
    const float* __restrict__ in, __bf16* __restrict__ out, int rows, int cols)
{
    __shared__ __bf16 tile[32][33];
    const int bx = blockIdx.x * 32;   // col offset
    const int by = blockIdx.y * 32;   // row offset
    const int tx = threadIdx.x & 31;
    const int ty = threadIdx.x >> 5;  // 0..7
#pragma unroll
    for (int r = 0; r < 32; r += 8)
        tile[ty + r][tx] = f2bf(in[(long long)(by + ty + r) * cols + (bx + tx)]);
    __syncthreads();
#pragma unroll
    for (int r = 0; r < 32; r += 8)
        out[(long long)(bx + ty + r) * rows + (by + tx)] = tile[tx][ty + r];
}

// ---------------------------------------------------------------------------
// Softmax stats over the QUERY axis n:  for each (b, m) column of S[b][n][m],
// compute running max c and Z^-1 = 1 / sum_n exp(S - c).  (online softmax)
// ---------------------------------------------------------------------------
__global__ void softmax_stats(const float* __restrict__ S,
                              float* __restrict__ cmax, float* __restrict__ zinv,
                              int Nn)
{
    const int b = blockIdx.y;
    const int m = blockIdx.x * blockDim.x + threadIdx.x;
    const float* Sb = S + (long long)b * Nn * Nn;
    float mx = -INFINITY, sum = 0.0f;
    for (int n = 0; n < Nn; ++n) {
        const float s  = Sb[(long long)n * Nn + m];
        const float nm = fmaxf(mx, s);
        sum = sum * __expf(mx - nm) + __expf(s - nm);
        mx  = nm;
    }
    cmax[(long long)b * Nn + m] = mx;
    zinv[(long long)b * Nn + m] = 1.0f / sum;
}

// P[b][n][m] = bf16(exp(S[b][n][m] - cmax[b][m]))
__global__ void make_p(const float* __restrict__ S, const float* __restrict__ cmax,
                       __bf16* __restrict__ P, int Nn)
{
    const int b = blockIdx.z;
    const int n = blockIdx.y;
    const int m = blockIdx.x * blockDim.x + threadIdx.x;
    const long long base = (long long)b * Nn * Nn;
    const float c = cmax[(long long)b * Nn + m];
    P[base + (long long)n * Nn + m] = f2bf(__expf(S[base + (long long)n * Nn + m] - c));
}

// xsT[b][d][m] = bf16(x[b][m][d] * zinv[b][m])   (scale + transpose, per batch)
__global__ __launch_bounds__(256) void scale_transpose(
    const float* __restrict__ x, const float* __restrict__ zinv,
    __bf16* __restrict__ out, int Nn, int Dd)
{
    const int b = blockIdx.z;
    const float* xb = x + (long long)b * Nn * Dd;
    const float* zb = zinv + (long long)b * Nn;
    __bf16* ob = out + (long long)b * Dd * Nn;

    __shared__ __bf16 tile[32][33];
    const int bx = blockIdx.x * 32;   // d offset
    const int by = blockIdx.y * 32;   // m offset
    const int tx = threadIdx.x & 31;
    const int ty = threadIdx.x >> 5;
#pragma unroll
    for (int r = 0; r < 32; r += 8) {
        const int m = by + ty + r;
        tile[ty + r][tx] = f2bf(xb[(long long)m * Dd + (bx + tx)] * zb[m]);
    }
    __syncthreads();
#pragma unroll
    for (int r = 0; r < 32; r += 8)
        ob[(long long)(bx + ty + r) * Nn + (by + tx)] = tile[tx][ty + r];
}

// ---------------------------------------------------------------------------
// Host-side launcher
// ---------------------------------------------------------------------------
extern "C" void kernel_launch(void* const* d_in, const int* in_sizes, int n_in,
                              void* d_out, int out_size, void* d_ws, size_t ws_size,
                              hipStream_t stream)
{
    (void)in_sizes; (void)n_in; (void)out_size; (void)ws_size;

    constexpr int B = 8, N = 2048, D = 1024, F = 512;
    constexpr long long MN = (long long)B * N;           // 16384 flattened rows

    const float* x  = (const float*)d_in[0];
    // d_in[1], d_in[2] = Wg, bg : dead branch, skipped
    const float* Wt = (const float*)d_in[3];
    const float* bt = (const float*)d_in[4];
    const float* Wp = (const float*)d_in[5];
    const float* bp = (const float*)d_in[6];
    float* out = (float*)d_out;

    // ---- workspace layout (bytes), everything written before read ----
    char* ws = (char*)d_ws;
    size_t off = 0;
    auto alloc = [&](size_t bytes) { void* p = ws + off; off = (off + bytes + 255) & ~(size_t)255; return p; };
    __bf16* xb    = (__bf16*)alloc(MN * D * 2);                 //  32 MB
    __bf16* thb   = (__bf16*)alloc(MN * F * 2);                 //  16 MB
    __bf16* phb   = (__bf16*)alloc(MN * F * 2);                 //  16 MB
    __bf16* xsT   = (__bf16*)alloc((long long)B * D * N * 2);   //  32 MB
    __bf16* WtT   = (__bf16*)alloc((long long)F * D * 2);       //   1 MB
    __bf16* WpT   = (__bf16*)alloc((long long)F * D * 2);       //   1 MB
    float*  cmax  = (float*)alloc((long long)B * N * 4);        //  64 KB
    float*  zinv  = (float*)alloc((long long)B * N * 4);        //  64 KB
    float*  S     = (float*)alloc((long long)B * N * N * 4);    // 128 MB
    __bf16* P     = (__bf16*)alloc((long long)B * N * N * 2);   //  64 MB

    // 1) x -> bf16
    f32_to_bf16<<<8192, 256, 0, stream>>>(x, xb, MN * D);
    // 2) weights -> bf16, transposed to [F, D]
    transpose_conv<<<dim3(F / 32, D / 32, 1), 256, 0, stream>>>(Wt, WtT, D, F);
    transpose_conv<<<dim3(F / 32, D / 32, 1), 256, 0, stream>>>(Wp, WpT, D, F);

    // 3) theta = sigmoid(x @ Wt + bt)  [MN x F]  (MODE 1)
    gemm_bf16_wmma<1><<<dim3(F / 256, (int)(MN / 128), 1), 256, 0, stream>>>(
        xb, WtT, nullptr, thb, bt, nullptr,
        (int)MN, F, D, 0, 0, 0, 0);
    // 4) phi = x @ Wp + bp  [MN x F]  (MODE 0)
    gemm_bf16_wmma<0><<<dim3(F / 256, (int)(MN / 128), 1), 256, 0, stream>>>(
        xb, WpT, nullptr, phb, bp, nullptr,
        (int)MN, F, D, 0, 0, 0, 0);

    // 5) S[b] = theta[b] @ phi[b]^T  [N x N] fp32  (MODE 2)
    gemm_bf16_wmma<2><<<dim3(N / 256, N / 128, B), 256, 0, stream>>>(
        thb, phb, S, nullptr, nullptr, nullptr,
        N, N, F,
        (long long)N * F, (long long)N * F, (long long)N * N, 0);

    // 6) per-column (query-axis) softmax stats
    softmax_stats<<<dim3(N / 256, B, 1), 256, 0, stream>>>(S, cmax, zinv, N);

    // 7) P = exp(S - cmax[m])  (bf16)
    make_p<<<dim3(N / 256, N, B), 256, 0, stream>>>(S, cmax, P, N);

    // 8) xsT[b][d][m] = x[b][m][d] * zinv[b][m]  (bf16, transposed)
    scale_transpose<<<dim3(D / 32, N / 32, B), 256, 0, stream>>>(x, zinv, xsT, N, D);

    // 9) out = P @ xsT^T + x   [N x D] fp32  (MODE 3)
    gemm_bf16_wmma<3><<<dim3(D / 256, N / 128, B), 256, 0, stream>>>(
        P, xsT, out, nullptr, nullptr, x,
        N, D, N,
        (long long)N * N, (long long)D * N, (long long)N * D, (long long)N * D);
}
